// DualBranchModel_88218628260045
// MI455X (gfx1250) — matrix operand order repaired and checked by reading, not compiled, not run on hardware
//
#include <hip/hip_runtime.h>
#include <hip/hip_bf16.h>
#include <math.h>

// ---------------- problem constants (from reference) ----------------
#define N_USERS   100000
#define N_ITEMS   50000
#define N_TOTAL   150000
#define IN_DIM    768
#define BRANCH    64
#define FUSED_D   128
#define N_EDGES   2097152
#define GCN_LAYERS 3
#define BATCH     4096
#define NUM_CAND  128
#define NEG       16
#define INV_TEMP  10.0f   // 1/TEMP, TEMP=0.1
#define KSTEPS    (IN_DIM / 32)          // 24
#define NTILES    (BRANCH / 16)          // 4

typedef __attribute__((ext_vector_type(16))) __bf16 bf16x16;
typedef __attribute__((ext_vector_type(8)))  float  f32x8;

__device__ __forceinline__ void atomic_add_f32(float* p, float v) {
  // lowers to global_atomic_add_f32 (hw FP atomic), no CAS loop
  __hip_atomic_fetch_add(p, v, __ATOMIC_RELAXED, __HIP_MEMORY_SCOPE_AGENT);
}

// ---------------- pack sem_w into WMMA B-fragment layout ----------------
// pack[((kstep*NTILES + ntile)*32 + lane)*16 + i] = bf16( W[K, N] )
//   K = kstep*32 + (i>>3)*16 + (lane>>4)*8 + (i&7)   (B layout: lanes 0-15 hold K {0..7,16..23}, 16-31 hold K {8..15,24..31})
//   N = ntile*16 + (lane&15)
__global__ void k_pack_b(const float* __restrict__ W, __bf16* __restrict__ pack) {
  int idx = blockIdx.x * blockDim.x + threadIdx.x;           // 24*4*32*16 = 49152
  if (idx >= KSTEPS * NTILES * 32 * 16) return;
  int i     = idx & 15;
  int lane  = (idx >> 4) & 31;
  int ntile = (idx >> 9) & 3;
  int kstep = idx >> 11;
  int K = kstep * 32 + (i >> 3) * 16 + (lane >> 4) * 8 + (i & 7);
  int N = ntile * 16 + (lane & 15);
  pack[idx] = (__bf16)W[(size_t)K * BRANCH + N];
}

// ---------------- semantic GEMM: out[M,64] = A[M,768] @ W[768,64] + b ----------------
// wave-per-16x64 tile; v_wmma_f32_16x16x32_bf16, 4 N-tiles, 24 K-steps.
// B fragments are pre-packed: one 32B vector load per fragment.
__global__ __launch_bounds__(256) void k_sem_gemm(
    const float* __restrict__ A, const bf16x16* __restrict__ Bp,
    const float* __restrict__ bias, float* __restrict__ out, int mtiles)
{
  const int lane   = threadIdx.x & 31;
  const int wid    = threadIdx.x >> 5;
  const int tile   = blockIdx.x * 8 + wid;
  if (tile >= mtiles) return;               // wave-uniform: EXEC stays all-ones for WMMA
  const int m0     = tile * 16;
  const int half   = lane >> 4;             // 0 or 1
  const int col    = lane & 15;
  const int row    = m0 + col;              // A-fragment: M = lane&15
  const int khalf  = half * 8;              // A-fragment K offset select

  f32x8 c0 = {}, c1 = {}, c2 = {}, c3 = {};
  const float* __restrict__ arow = A + (size_t)row * IN_DIM;

#pragma unroll 2
  for (int kstep = 0; kstep < KSTEPS; ++kstep) {
    const int kk = kstep * 32;
    // ---- A fragment (16-bit A layout: lane<16 -> K {0..7,16..23}; lane>=16 -> K {8..15,24..31})
    const float4* p0 = reinterpret_cast<const float4*>(arow + kk + khalf);
    const float4* p1 = reinterpret_cast<const float4*>(arow + kk + 16 + khalf);
    float4 x0 = p0[0], x1 = p0[1];
    float4 y0 = p1[0], y1 = p1[1];
    bf16x16 a;
    a[0]=(__bf16)x0.x; a[1]=(__bf16)x0.y; a[2]=(__bf16)x0.z; a[3]=(__bf16)x0.w;
    a[4]=(__bf16)x1.x; a[5]=(__bf16)x1.y; a[6]=(__bf16)x1.z; a[7]=(__bf16)x1.w;
    a[8]=(__bf16)y0.x; a[9]=(__bf16)y0.y; a[10]=(__bf16)y0.z; a[11]=(__bf16)y0.w;
    a[12]=(__bf16)y1.x; a[13]=(__bf16)y1.y; a[14]=(__bf16)y1.z; a[15]=(__bf16)y1.w;

    // ---- B fragments: pre-packed, L2-resident; one 32B load each
    const bf16x16* pv = Bp + (size_t)kstep * (NTILES * 32) + lane;
    bf16x16 b0 = pv[0 * 32];
    bf16x16 b1 = pv[1 * 32];
    bf16x16 b2 = pv[2 * 32];
    bf16x16 b3 = pv[3 * 32];

    c0 = __builtin_amdgcn_wmma_f32_16x16x32_bf16(false, a, false, b0, (short)0, c0, false, false);
    c1 = __builtin_amdgcn_wmma_f32_16x16x32_bf16(false, a, false, b1, (short)0, c1, false, false);
    c2 = __builtin_amdgcn_wmma_f32_16x16x32_bf16(false, a, false, b2, (short)0, c2, false, false);
    c3 = __builtin_amdgcn_wmma_f32_16x16x32_bf16(false, a, false, b3, (short)0, c3, false, false);
  }

  // ---- C/D layout: vgpr r -> M = r + half*8, N = lane&15
  const int rbase = m0 + half * 8;
#pragma unroll
  for (int r = 0; r < 8; ++r) {
    float* o = out + (size_t)(rbase + r) * BRANCH + col;
    o[0]  = c0[r] + bias[col];
    o[16] = c1[r] + bias[16 + col];
    o[32] = c2[r] + bias[32 + col];
    o[48] = c3[r] + bias[48 + col];
  }
}

// ---------------- GCN helpers ----------------
__global__ void k_init_ego(const float* __restrict__ uw, const float* __restrict__ iw,
                           float* __restrict__ ego, float* __restrict__ acc) {
  size_t i = (size_t)blockIdx.x * blockDim.x + threadIdx.x;
  const size_t nu = (size_t)N_USERS * BRANCH;
  const size_t nt = (size_t)N_TOTAL * BRANCH;
  if (i >= nt) return;
  float v = (i < nu) ? uw[i] : iw[i - nu];
  ego[i] = v; acc[i] = v;
}

__global__ void k_zero(float* __restrict__ p, size_t n) {
  size_t i = (size_t)blockIdx.x * blockDim.x + threadIdx.x;
  if (i < n) p[i] = 0.0f;
}

// edge x lane parallel scatter: out[row] += val * ego[col]; 2 floats per lane
__global__ void k_scatter(const float* __restrict__ vals, const int* __restrict__ rows,
                          const int* __restrict__ cols, const float* __restrict__ ego,
                          float* __restrict__ outego, int nedges) {
  long long tid = (long long)blockIdx.x * blockDim.x + threadIdx.x;
  int e    = (int)(tid >> 5);
  int lane = (int)(tid & 31);
  if (e >= nedges) return;
  int r = rows[e], c = cols[e];
  float v = vals[e];
  const float* src = ego    + (size_t)c * BRANCH + lane * 2;
  float*       dst = outego + (size_t)r * BRANCH + lane * 2;
  atomic_add_f32(dst + 0, v * src[0]);
  atomic_add_f32(dst + 1, v * src[1]);
}

__global__ void k_axpy(float* __restrict__ acc, const float* __restrict__ x, size_t n) {
  size_t i = (size_t)blockIdx.x * blockDim.x + threadIdx.x;
  if (i < n) acc[i] += x[i];
}

// ---------------- fuse: out[row,0:64]=norm(acc/4), out[row,64:128]=norm(sem); renorm concat ----------------
__global__ __launch_bounds__(256) void k_fuse(const float* __restrict__ collab,
                                              const float* __restrict__ sem,
                                              float* __restrict__ out, int rows) {
  int gw   = (int)(((size_t)blockIdx.x * blockDim.x + threadIdx.x) >> 5);
  int lane = threadIdx.x & 31;
  if (gw >= rows) return;
  size_t base = (size_t)gw * BRANCH + lane * 2;
  float c0 = collab[base] * 0.25f;      // mean over (GCN_LAYERS+1) stacked layers
  float c1 = collab[base + 1] * 0.25f;
  float s0 = sem[base];
  float s1 = sem[base + 1];
  float nc = c0 * c0 + c1 * c1;
  float ns = s0 * s0 + s1 * s1;
#pragma unroll
  for (int off = 16; off > 0; off >>= 1) {
    nc += __shfl_xor(nc, off, 32);
    ns += __shfl_xor(ns, off, 32);
  }
  nc = fmaxf(sqrtf(nc), 1e-12f);
  ns = fmaxf(sqrtf(ns), 1e-12f);
  float h0 = c0 / nc, h1 = c1 / nc, g0 = s0 / ns, g1 = s1 / ns;
  float tt = h0 * h0 + h1 * h1 + g0 * g0 + g1 * g1;
#pragma unroll
  for (int off = 16; off > 0; off >>= 1) tt += __shfl_xor(tt, off, 32);
  float nf = fmaxf(sqrtf(tt), 1e-12f);
  float* o = out + (size_t)gw * FUSED_D;
  o[lane * 2]          = h0 / nf;
  o[lane * 2 + 1]      = h1 / nf;
  o[64 + lane * 2]     = g0 / nf;
  o[64 + lane * 2 + 1] = g1 / nf;
}

// ---------------- candidate + positive scoring (block per batch row) ----------------
__global__ __launch_bounds__(128) void k_score(const float* __restrict__ u_all,
                                               const float* __restrict__ i_all,
                                               const int* __restrict__ uids,
                                               const int* __restrict__ pos_iids,
                                               const int* __restrict__ cand_ids,
                                               float* __restrict__ scores,
                                               float* __restrict__ pos_scores) {
  __shared__ float u[FUSED_D];
  int b = blockIdx.x, t = threadIdx.x;
  int uid = uids[b];
  u[t] = u_all[(size_t)uid * FUSED_D + t];
  __syncthreads();
  int pos = pos_iids[b];
  int cid = cand_ids[(size_t)b * NUM_CAND + t];
  if (cid == pos) cid = (cid + 1) % N_ITEMS;        // collision fix, as in reference
  const float* e = i_all + (size_t)cid * FUSED_D;
  float acc = 0.0f;
#pragma unroll 8
  for (int d = 0; d < FUSED_D; ++d) acc += e[d] * u[d];
  scores[(size_t)b * NUM_CAND + t] = acc;
  if (t == 0) {
    const float* p = i_all + (size_t)pos * FUSED_D;
    float a2 = 0.0f;
    for (int d = 0; d < FUSED_D; ++d) a2 += p[d] * u[d];
    pos_scores[b] = a2;
  }
}

// ---------------- top-16 selection + InfoNCE loss (thread per row) ----------------
__global__ void k_topk_loss(const float* __restrict__ scores,
                            const float* __restrict__ pos_scores,
                            float* __restrict__ out) {
  int b = blockIdx.x * blockDim.x + threadIdx.x;
  if (b >= BATCH) return;
  const float* s = scores + (size_t)b * NUM_CAND;
  unsigned sel[4] = {0u, 0u, 0u, 0u};
  float neg[NEG];
#pragma unroll 1
  for (int k = 0; k < NEG; ++k) {
    float best = -1e30f; int bi = 0;
    for (int c = 0; c < NUM_CAND; ++c) {
      if ((sel[c >> 5] >> (c & 31)) & 1u) continue;
      float v = s[c];
      if (v > best) { best = v; bi = c; }
    }
    sel[bi >> 5] |= 1u << (bi & 31);
    neg[k] = best * INV_TEMP;
  }
  float pos = pos_scores[b] * INV_TEMP;
  float m = pos;
#pragma unroll
  for (int k = 0; k < NEG; ++k) m = fmaxf(m, neg[k]);
  float se = __expf(pos - m);
#pragma unroll
  for (int k = 0; k < NEG; ++k) se += __expf(neg[k] - m);
  float loss = -(pos - (m + __logf(se)));
  atomic_add_f32(out, loss * (1.0f / (float)BATCH));
}

__global__ void k_init_out(float* out) { out[0] = 0.0f; }

// ---------------- host orchestration ----------------
extern "C" void kernel_launch(void* const* d_in, const int* in_sizes, int n_in,
                              void* d_out, int out_size, void* d_ws, size_t ws_size,
                              hipStream_t stream) {
  const float* raw_item_embs = (const float*)d_in[0];   // [50000,768]
  const float* user_sem_base = (const float*)d_in[1];   // [100000,768]
  const float* sem_w         = (const float*)d_in[2];   // [768,64]
  const float* sem_b         = (const float*)d_in[3];   // [64]
  const float* collab_user_w = (const float*)d_in[4];   // [100000,64]
  const float* collab_item_w = (const float*)d_in[5];   // [50000,64]
  const float* adj_vals      = (const float*)d_in[6];   // [E]
  const int*   adj_rows      = (const int*)d_in[7];
  const int*   adj_cols      = (const int*)d_in[8];
  const int*   uids          = (const int*)d_in[9];
  const int*   pos_iids      = (const int*)d_in[10];
  const int*   cand_ids      = (const int*)d_in[11];
  float* out = (float*)d_out;

  // ---- workspace carving (~235 MB) ----
  char* w = (char*)d_ws;
  size_t off = 0;
  float* u_sem  = (float*)(w + off); off += (size_t)N_USERS * BRANCH * 4;      // 25.6 MB
  float* i_sem  = (float*)(w + off); off += (size_t)N_ITEMS * BRANCH * 4;      // 12.8 MB
  float* egoA   = (float*)(w + off); off += (size_t)N_TOTAL * BRANCH * 4;      // 38.4 MB
  float* egoB   = (float*)(w + off); off += (size_t)N_TOTAL * BRANCH * 4;      // 38.4 MB
  float* acc    = (float*)(w + off); off += (size_t)N_TOTAL * BRANCH * 4;      // 38.4 MB
  float* u_all  = (float*)(w + off); off += (size_t)N_USERS * FUSED_D * 4;     // 51.2 MB
  float* i_all  = (float*)(w + off); off += (size_t)N_ITEMS * FUSED_D * 4;     // 25.6 MB
  __bf16* Bp    = (__bf16*)(w + off); off += (size_t)KSTEPS * NTILES * 32 * 16 * 2; // 96 KB
  float* scores = (float*)(w + off); off += (size_t)BATCH * NUM_CAND * 4;      // 2 MB
  float* posS   = (float*)(w + off); off += (size_t)BATCH * 4;                 // 16 KB
  (void)ws_size; (void)in_sizes; (void)n_in; (void)out_size;

  // 1) pack weights into WMMA B-fragment layout (bf16)
  k_pack_b<<<(KSTEPS * NTILES * 32 * 16 + 255) / 256, 256, 0, stream>>>(sem_w, Bp);

  // 2) semantic GEMMs via WMMA (users: 6250 tiles, items: 3125 tiles)
  k_sem_gemm<<<(6250 + 7) / 8, 256, 0, stream>>>(user_sem_base, (const bf16x16*)Bp, sem_b, u_sem, 6250);
  k_sem_gemm<<<(3125 + 7) / 8, 256, 0, stream>>>(raw_item_embs, (const bf16x16*)Bp, sem_b, i_sem, 3125);

  // 3) LightGCN propagation (3 layers, ping-pong ego buffers)
  const size_t totElems = (size_t)N_TOTAL * BRANCH;                  // 9.6M
  const int blocksTot = (int)((totElems + 255) / 256);
  k_init_ego<<<blocksTot, 256, 0, stream>>>(collab_user_w, collab_item_w, egoA, acc);
  const int scatterBlocks = (int)(((long long)N_EDGES * 32 + 255) / 256);
  float* src = egoA; float* dst = egoB;
  for (int l = 0; l < GCN_LAYERS; ++l) {
    k_zero<<<blocksTot, 256, 0, stream>>>(dst, totElems);
    k_scatter<<<scatterBlocks, 256, 0, stream>>>(adj_vals, adj_rows, adj_cols, src, dst, N_EDGES);
    k_axpy<<<blocksTot, 256, 0, stream>>>(acc, dst, totElems);
    float* t = src; src = dst; dst = t;
  }

  // 4) norm-concat-norm fusion
  k_fuse<<<(N_USERS * 32) / 256, 256, 0, stream>>>(acc, u_sem, u_all, N_USERS);
  k_fuse<<<(N_ITEMS * 32) / 256, 256, 0, stream>>>(acc + (size_t)N_USERS * BRANCH, i_sem, i_all, N_ITEMS);

  // 5) candidate scoring + positive logits
  k_score<<<BATCH, 128, 0, stream>>>(u_all, i_all, uids, pos_iids, cand_ids, scores, posS);

  // 6) top-16 hard negatives (== top scores) + softmax-CE loss, mean-reduced into d_out
  k_init_out<<<1, 1, 0, stream>>>(out);
  k_topk_loss<<<BATCH / 256, 256, 0, stream>>>(scores, posS, out);
}
